// BiGRUEncoder_90821378441663
// MI455X (gfx1250) — matrix operand order repaired and checked by reading, not compiled, not run on hardware
//
#include <hip/hip_runtime.h>
#include <hip/hip_bf16.h>

#define TT 2000
#define NB 64
#define DIN 512
#define HH 256

typedef __attribute__((ext_vector_type(16))) __bf16        v16bf;
typedef __attribute__((ext_vector_type(8)))  float         v8f;
typedef __attribute__((ext_vector_type(4)))  unsigned int  v4u;
typedef __attribute__((ext_vector_type(4)))  int           v4i;
typedef __attribute__((ext_vector_type(8)))  int           v8i;

#if __has_builtin(__builtin_amdgcn_tensor_load_to_lds)
#define HAVE_TDM 1
#else
#define HAVE_TDM 0
#endif

struct Pair32 { v4u lo, hi; };

__device__ __forceinline__ v16bf mk_frag(v4u lo, v4u hi) {
  Pair32 p{lo, hi};
  return __builtin_bit_cast(v16bf, p);
}

__device__ __forceinline__ unsigned short f2bf(float f) {
  unsigned u = __builtin_bit_cast(unsigned, f);
  unsigned r = u + 0x7FFFu + ((u >> 16) & 1u);   // round-to-nearest-even
  return (unsigned short)(r >> 16);
}
__device__ __forceinline__ float bf2f(unsigned short h) {
  unsigned u = ((unsigned)h) << 16;
  return __builtin_bit_cast(float, u);
}
__device__ __forceinline__ float sigmoidf(float x) {
  return 1.0f / (1.0f + __expf(-x));
}

// A fragment (16x32 bf16, M x K): lane<16 -> row l, K {k0..k0+7, k0+16..k0+23};
// lane>=16 -> row l-16, K halves shifted by 8.  Two 16B loads.
__device__ __forceinline__ v16bf load_a_frag(const unsigned short* base, int stride,
                                             int row0, int k0, int lane) {
  int l = lane & 15, hi = lane >> 4;
  const unsigned short* p = base + (size_t)(row0 + l) * stride + k0 + hi * 8;
  v4u x = *(const v4u*)p;
  v4u y = *(const v4u*)(p + 16);
  return mk_frag(x, y);
}
// B fragment (32x16 bf16, K x N) from row-major W (N rows, K cols):
// lane%16 = output column (W row), K {k0..k0+7, k0+16..k0+23}, lane>=16 shifted by 8.
__device__ __forceinline__ v16bf load_b_frag(const unsigned short* base, int stride,
                                             int col0, int k0, int lane) {
  int l = lane & 15, hi = lane >> 4;
  const unsigned short* p = base + (size_t)(col0 + l) * stride + k0 + hi * 8;
  v4u x = *(const v4u*)p;
  v4u y = *(const v4u*)(p + 16);
  return mk_frag(x, y);
}

__device__ __forceinline__ v8f wmma_bf16(v16bf a, v16bf b, v8f c) {
  return __builtin_amdgcn_wmma_f32_16x16x32_bf16(false, a, false, b, (short)0, c,
                                                 false, false);
}

__device__ __forceinline__ void wait_tensorcnt0() {
#if __has_builtin(__builtin_amdgcn_s_wait_tensorcnt)
  __builtin_amdgcn_s_wait_tensorcnt(0);
#else
  asm volatile("s_wait_tensorcnt 0x0" ::: "memory");
#endif
}

#if HAVE_TDM
// 2D TDM tile load: rows x cols bf16 tile, row stride in elements, into LDS.
// 6-arg toolchain form: (g0:u32x4, g1:i32x8, g2:i32x4, g3:i32x4, g4:i32x8, cpol)
__device__ __forceinline__ void tdm_load_tile_2d(unsigned lds_byte_off,
                                                 const unsigned short* gsrc,
                                                 unsigned rows, unsigned cols,
                                                 unsigned row_stride) {
  unsigned long long ga = (unsigned long long)gsrc;
  v4u g0 = { 1u,                                   // count=1, user descriptor
             lds_byte_off,
             (unsigned)ga,
             ((unsigned)(ga >> 32) & 0x01FFFFFFu) | 0x80000000u };  // type=2
  v8i g1 = { (int)(1u << 16),                                  // data_size=1 (2B)
             (int)((cols & 0xFFFFu) << 16),                    // tensor_dim0 lo
             (int)(((cols >> 16) & 0xFFFFu) | ((rows & 0xFFFFu) << 16)),
             (int)(((rows >> 16) & 0xFFFFu) | ((cols & 0xFFFFu) << 16)), // tile_dim0
             (int)(rows & 0xFFFFu),                            // tile_dim1
             (int)row_stride,                                  // dim0 stride lo
             0, 0 };
  v4i g2 = {0, 0, 0, 0};
  v4i g3 = {0, 0, 0, 0};
  v8i g4 = {0, 0, 0, 0, 0, 0, 0, 0};
  __builtin_amdgcn_tensor_load_to_lds(g0, g1, g2, g3, g4, 0);
}
#endif

// Device-scope spin barrier between the 4 WGs of one direction.
__device__ __forceinline__ void grid4_barrier(unsigned* ctr, unsigned target) {
  __syncthreads();
  if (threadIdx.x == 0) {
    __threadfence();                       // release stores of h slice
    atomicAdd(ctr, 1u);
    while (*((volatile unsigned*)ctr) < target) { __builtin_amdgcn_s_sleep(1); }
    __threadfence();                       // acquire / invalidate WGP$
  }
  __syncthreads();
}

struct LayerArgs {
  const float* x_f32;             // layer0 input (T,N,512) fp32, or null
  const unsigned short* x_bf;     // layer1 input (T,N,512) bf16, or null
  const unsigned short* wih0; const unsigned short* wih1;   // bf16 (768 x Din)
  const unsigned short* whh0; const unsigned short* whh1;   // bf16 (768 x 256)
  const float* bih0; const float* bih1;
  const float* bhh0; const float* bhh1;
  unsigned short* y_bf;           // layer0 output (bf16) or null
  float* y_f32;                   // layer1 output (fp32, d_out) or null
  unsigned short* hglob;          // [dir][pingpong][64*256] bf16
  unsigned* sync;                 // [dir] spin counters
};

__global__ __launch_bounds__(256, 1) void bigru_layer_kernel(LayerArgs a) {
  const int tid  = threadIdx.x;
  const int lane = tid & 31, wave = tid >> 5;
  const int dir  = blockIdx.x >> 2;     // 0 = forward, 1 = backward
  const int c    = blockIdx.x & 3;      // hidden-column chunk (64 cols)
  const int mtile = wave >> 1, par = wave & 1;

  extern __shared__ unsigned short smem[];
  unsigned short* s_whh = smem;                      // 192x256 bf16 = 96 KB
  unsigned short* s_x   = smem + 49152;              // 64x512 bf16 = 64 KB (phase A)
  float* s_scr  = (float*)(smem + 49152);            // 64x256 f32 = 64 KB (aliased)
  float* s_bias = (float*)(smem + 49152 + 32768);    // 256 f32 = 1 KB

  const unsigned short* wih = dir ? a.wih1 : a.wih0;
  const unsigned short* whh = dir ? a.whh1 : a.whh0;
  const float* bih = dir ? a.bih1 : a.bih0;
  const float* bhh = dir ? a.bhh1 : a.bhh0;
  unsigned* ctr = a.sync + dir;
  unsigned short* hbase = a.hglob + (size_t)dir * 2 * (NB * HH);

  // ---- init: W_hh slice -> LDS via TDM (3 tiles of 64x256 bf16) ----
#if HAVE_TDM
  if (wave == 0) {
    unsigned lds0 = (unsigned)(unsigned long long)(const void*)s_whh;
#pragma unroll
    for (int g = 0; g < 3; ++g)
      tdm_load_tile_2d(lds0 + (unsigned)g * 32768u,
                       whh + (size_t)(g * 256 + c * 64) * 256, 64u, 256u, 256u);
  }
  wait_tensorcnt0();
#else
  for (int i = tid; i < 192 * 32; i += 256) {        // 32 x v4u per slice row
    int s = i >> 5, q = i & 31;
    int orig = (s >> 6) * 256 + c * 64 + (s & 63);
    *((v4u*)(s_whh + s * 256 + q * 8)) = *((const v4u*)(whh + (size_t)orig * 256 + q * 8));
  }
#endif

  if (tid < 64) {                                    // per-chunk gate biases
    int col = c * 64 + tid;
    s_bias[tid]       = bih[col]       + bhh[col];         // r
    s_bias[64 + tid]  = bih[256 + col] + bhh[256 + col];   // z
    s_bias[128 + tid] = bih[512 + col];                    // n (input side)
    s_bias[192 + tid] = bhh[512 + col];                    // n (hidden side)
  }

  // zero our chunk of the h buffer read at t=0 (pingpong buf 1)
  for (int i = tid; i < 64 * 64; i += 256) {
    int row = i >> 6, j = i & 63;
    hbase[(NB * HH) + row * HH + c * 64 + j] = 0;
  }

  unsigned epoch = 0;
  grid4_barrier(ctr, 4u * (++epoch));

  const v8f vzero = {0.f, 0.f, 0.f, 0.f, 0.f, 0.f, 0.f, 0.f};

  for (int t = 0; t < TT; ++t) {
    const int trow = dir ? (TT - 1 - t) : t;
    const unsigned short* hprev = hbase + ((t + 1) & 1) * (NB * HH);
    unsigned short* hcur        = hbase + (t & 1) * (NB * HH);

    // ---- stage x_t into LDS as bf16 ----
    if (a.x_f32) {
      const float4* xs4 = (const float4*)(a.x_f32 + (size_t)trow * NB * DIN);
      unsigned int* dst = (unsigned int*)s_x;
      for (int i = tid; i < (NB * DIN) / 4; i += 256) {
        float4 f = xs4[i];
        dst[2 * i]     = ((unsigned)f2bf(f.y) << 16) | f2bf(f.x);
        dst[2 * i + 1] = ((unsigned)f2bf(f.w) << 16) | f2bf(f.z);
      }
    } else {
      const v4u* src = (const v4u*)(a.x_bf + (size_t)trow * NB * DIN);
      v4u* dst = (v4u*)s_x;
      for (int i = tid; i < (NB * DIN) / 8; i += 256) dst[i] = src[i];
    }
    __syncthreads();

    v8f accRZ[4] = {vzero, vzero, vzero, vzero};   // combined gx+gh for r,z tiles
    v8f accNX[2] = {vzero, vzero};                 // gx for n tiles
    v8f accNH[2] = {vzero, vzero};                 // gh for n tiles

    // ---- phase A: gx slice = x_t * W_ih_slice^T  (K = 512) ----
    for (int kk = 0; kk < 16; ++kk) {
      v16bf af = load_a_frag(s_x, DIN, mtile * 16, kk * 32, lane);
      if (kk + 1 < 16)
        __builtin_prefetch(wih + (size_t)(c * 64 + (lane & 15)) * DIN + (kk + 1) * 32, 0, 1);
#pragma unroll
      for (int i = 0; i < 6; ++i) {
        int nt = 2 * i + par;
        int orig = (nt >> 2) * 256 + c * 64 + (nt & 3) * 16;   // gate-block row
        v16bf bfr = load_b_frag(wih, DIN, orig, kk * 32, lane);
        if (i < 4) accRZ[i]     = wmma_bf16(af, bfr, accRZ[i]);
        else       accNX[i - 4] = wmma_bf16(af, bfr, accNX[i - 4]);
      }
    }
    // ---- phase B: gh slice = h_{t-1} * W_hh_slice^T  (K = 256) ----
    for (int kk = 0; kk < 8; ++kk) {
      v16bf af = load_a_frag(hprev, HH, mtile * 16, kk * 32, lane);
#pragma unroll
      for (int i = 0; i < 6; ++i) {
        int nt = 2 * i + par;
        v16bf bfr = load_b_frag(s_whh, HH, nt * 16, kk * 32, lane);
        if (i < 4) accRZ[i]     = wmma_bf16(af, bfr, accRZ[i]);
        else       accNH[i - 4] = wmma_bf16(af, bfr, accNH[i - 4]);
      }
    }

    __syncthreads();            // all x reads done; scratch aliases s_x
    {
      int l15 = lane & 15, hi = lane >> 4;
#pragma unroll
      for (int r = 0; r < 8; ++r) {
        int row = mtile * 16 + hi * 8 + r;
#pragma unroll
        for (int i = 0; i < 4; ++i) {
          int col = (2 * i + par) * 16 + l15;           // [0,128): r then z
          s_scr[row * 256 + col] = accRZ[i][r];
        }
#pragma unroll
        for (int i = 0; i < 2; ++i) {
          int col = (2 * i + par) * 16 + l15;           // [0,64) within n block
          s_scr[row * 256 + 128 + col] = accNX[i][r];
          s_scr[row * 256 + 192 + col] = accNH[i][r];
        }
      }
    }
    __syncthreads();

    // ---- gates + state update for our 64 hidden columns ----
    for (int idx = tid; idx < 64 * 64; idx += 256) {
      int row = idx >> 6, j = idx & 63;
      float vr  = s_scr[row * 256 + j];
      float vz  = s_scr[row * 256 + 64 + j];
      float gxn = s_scr[row * 256 + 128 + j];
      float ghn = s_scr[row * 256 + 192 + j];
      float r = sigmoidf(vr + s_bias[j]);
      float z = sigmoidf(vz + s_bias[64 + j]);
      float n = tanhf(gxn + s_bias[128 + j] + r * (ghn + s_bias[192 + j]));
      int col = c * 64 + j;
      float hp = bf2f(hprev[row * HH + col]);
      float hn = (1.0f - z) * n + z * hp;
      hcur[row * HH + col] = f2bf(hn);
      size_t yoff = ((size_t)trow * NB + row) * (2 * HH) + (size_t)dir * HH + col;
      if (a.y_f32) a.y_f32[yoff] = hn;
      else         a.y_bf[yoff]  = f2bf(hn);
    }

    grid4_barrier(ctr, 4u * (++epoch));   // h_t visible to all 4 WGs
  }
}

__global__ void convert_f32_bf16_kernel(const float* __restrict__ s,
                                        unsigned short* __restrict__ d, int n) {
  int i = blockIdx.x * blockDim.x + threadIdx.x;
  if (i < n) d[i] = f2bf(s[i]);
}

extern "C" void kernel_launch(void* const* d_in, const int* in_sizes, int n_in,
                              void* d_out, int out_size, void* d_ws, size_t ws_size,
                              hipStream_t stream) {
  (void)in_sizes; (void)n_in; (void)out_size; (void)ws_size;

  // workspace layout (element offsets in unsigned short)
  unsigned short* W = (unsigned short*)d_ws;
  unsigned short* wih_bf[4] = { W, W + 393216, W + 786432, W + 1179648 };    // l0f,l0b,l1f,l1b
  unsigned short* whh_bf[4] = { W + 1572864, W + 1769472, W + 1966080, W + 2162688 };
  unsigned short* y0    = W + 2359296;                 // (T,N,512) bf16: 65,536,000
  unsigned short* hglob = W + 2359296 + 65536000;      // 4 x 16384 bf16
  unsigned* sync = (unsigned*)(hglob + 65536);

  const int wih_idx[4] = {1, 5, 9, 13};
  const int whh_idx[4] = {2, 6, 10, 14};
  for (int i = 0; i < 4; ++i) {
    int n1 = 768 * 512;
    convert_f32_bf16_kernel<<<(n1 + 255) / 256, 256, 0, stream>>>(
        (const float*)d_in[wih_idx[i]], wih_bf[i], n1);
    int n2 = 768 * 256;
    convert_f32_bf16_kernel<<<(n2 + 255) / 256, 256, 0, stream>>>(
        (const float*)d_in[whh_idx[i]], whh_bf[i], n2);
  }

  (void)hipFuncSetAttribute((const void*)bigru_layer_kernel,
                            hipFuncAttributeMaxDynamicSharedMemorySize, 164864);

  // ---- layer 0 ----
  (void)hipMemsetAsync(sync, 0, 64, stream);
  LayerArgs a0{};
  a0.x_f32 = (const float*)d_in[0]; a0.x_bf = nullptr;
  a0.wih0 = wih_bf[0]; a0.wih1 = wih_bf[1];
  a0.whh0 = whh_bf[0]; a0.whh1 = whh_bf[1];
  a0.bih0 = (const float*)d_in[3];  a0.bhh0 = (const float*)d_in[4];
  a0.bih1 = (const float*)d_in[7];  a0.bhh1 = (const float*)d_in[8];
  a0.y_bf = y0; a0.y_f32 = nullptr;
  a0.hglob = hglob; a0.sync = sync;
  bigru_layer_kernel<<<dim3(8), dim3(256), 164864, stream>>>(a0);

  // ---- layer 1 ----
  (void)hipMemsetAsync(sync, 0, 64, stream);
  LayerArgs a1{};
  a1.x_f32 = nullptr; a1.x_bf = y0;
  a1.wih0 = wih_bf[2]; a1.wih1 = wih_bf[3];
  a1.whh0 = whh_bf[2]; a1.whh1 = whh_bf[3];
  a1.bih0 = (const float*)d_in[11]; a1.bhh0 = (const float*)d_in[12];
  a1.bih1 = (const float*)d_in[15]; a1.bhh1 = (const float*)d_in[16];
  a1.y_bf = nullptr; a1.y_f32 = (float*)d_out;
  a1.hglob = hglob; a1.sync = sync;
  bigru_layer_kernel<<<dim3(8), dim3(256), 164864, stream>>>(a1);
}
